// attentionDecoderRNN_88046829568826
// MI455X (gfx1250) — matrix operand order repaired and checked by reading, not compiled, not run on hardware
//
#include <hip/hip_runtime.h>

typedef __attribute__((ext_vector_type(16))) __bf16 v16bf;
typedef __attribute__((ext_vector_type(8)))  float  v8f;

static constexpr int kH = 1024, kE = 1024, kA = 100, kV = 50257, kL = 512, kS = 1024;
static constexpr int kG = 64;   // scan grid size (blocks); slices: H/64=16, S/64=16

// ---------------------------------------------------------------- utilities
__device__ __forceinline__ float warp_sum(float v) {
#pragma unroll
  for (int o = 16; o > 0; o >>= 1) v += __shfl_down(v, o, 32);
  return v;
}

__device__ __forceinline__ float dot4(float4 a, float4 b) {
  return a.x * b.x + a.y * b.y + a.z * b.z + a.w * b.w;
}

__device__ __forceinline__ void grid_barrier(unsigned* cnt, unsigned& epoch) {
  __threadfence();
  __syncthreads();
  if (threadIdx.x == 0) {
    atomicAdd(cnt, 1u);
    epoch += gridDim.x;
    while (*(volatile unsigned*)cnt < epoch) __builtin_amdgcn_s_sleep(2);
  }
  __syncthreads();
  __threadfence();
}

// ---------------------------------------------------------------- prep
__global__ void k_embed(const int* __restrict__ tok, const float* __restrict__ emb,
                        float* __restrict__ out) {
  const int row = blockIdx.x;
  const int t = tok[row];
  const float4* src = (const float4*)(emb + (size_t)t * kE);
  float4* dst = (float4*)(out + (size_t)row * kE);
  for (int j = threadIdx.x; j < kE / 4; j += 256) dst[j] = src[j];
}

// out[r][a] = dot(X[r,:K], W[a,:K]) + b[a]
__global__ void k_proj(const float* __restrict__ X, const float* __restrict__ W,
                       const float* __restrict__ b, float* __restrict__ out, int K) {
  const int r = blockIdx.x;
  const int wv = threadIdx.x >> 5, lane = threadIdx.x & 31;
  const float4* xr = (const float4*)(X + (size_t)r * K);
  for (int a = wv; a < kA; a += 8) {
    const float4* wa = (const float4*)(W + (size_t)a * K);
    float acc = 0.f;
    for (int k = lane; k < K / 4; k += 32) acc += dot4(xr[k], wa[k]);
    acc = warp_sum(acc);
    if (lane == 0) out[r * kA + a] = acc + b[a];
  }
}

// ---------------------------------------------------------------- sequential scan
// 64 blocks x 256 threads persistent; only TWO grid barriers per step:
// attention state (ah1, logits, softmax) is recomputed redundantly per block from
// L2-resident data; barriers remain only for ctx slices and the h update.
__global__ void k_scan(const float* __restrict__ embedded, const float* __restrict__ enc,
                       const float* __restrict__ hidden0,
                       const float* __restrict__ W1,  const float* __restrict__ b1,
                       const float* __restrict__ Wa,  const float* __restrict__ ba,
                       const float* __restrict__ Wih, const float* __restrict__ bih,
                       const float* __restrict__ Whh, const float* __restrict__ bhh,
                       const float* __restrict__ ah2, const float* __restrict__ ah3,
                       float* hbuf, float* ctx, float* Hall,
                       float* out_attn, float* out_hfinal, unsigned* bar) {
  const int blk = blockIdx.x, tid = threadIdx.x;
  const int wv = tid >> 5, lane = tid & 31;
  unsigned epoch = 0;

  __shared__ float  s_ah1[112];
  __shared__ float  s_wa[112];
  __shared__ float  s_ah3t[112];
  __shared__ float  s_logit[kS];
  __shared__ float  s_attn[kS];
  __shared__ float  red[256];
  __shared__ float4 red4[256];
  __shared__ float  s_gi[48], s_gh[48];

  if (blk == 0)
    for (int j = tid; j < kH; j += 256) hbuf[j] = hidden0[j];
  if (tid < kA) s_wa[tid] = Wa[tid];
  const float ba0 = ba[0];
  grid_barrier(bar, epoch);

  for (int t = 0; t < kL; ++t) {
    const float* hcur  = hbuf + (t & 1) * kH;
    float*       hnext = hbuf + ((t + 1) & 1) * kH;
    const float4* h4   = (const float4*)hcur;
    const float4* emb4 = (const float4*)(embedded + (size_t)t * kE);

    if (tid < kA) s_ah3t[tid] = ah3[t * kA + tid];

    // S1: ah1 = W1 @ h + b1, redundant per block (W1 is L2-resident)
    for (int a = wv; a < kA; a += 8) {
      const float4* w4 = (const float4*)(W1 + (size_t)a * kH);
      float acc = 0.f;
      for (int k = lane; k < kH / 4; k += 32) acc += dot4(w4[k], h4[k]);
      acc = warp_sum(acc);
      if (lane == 0) s_ah1[a] = acc + b1[a];
    }
    __syncthreads();

    // S2: full logits redundant per block (ah2 is L2-resident)
    for (int s = tid; s < kS; s += 256) {
      const float* a2 = ah2 + s * kA;
      float acc = 0.f;
      for (int a = 0; a < kA; ++a) {
        float e = s_ah1[a] + a2[a] + s_ah3t[a];
        e = e > 0.f ? e : 0.f;
        acc += e * s_wa[a];
      }
      s_logit[s] = acc + ba0;
    }
    __syncthreads();

    // S3: block-local softmax
    float m = -3.0e38f;
    for (int s = tid; s < kS; s += 256) m = fmaxf(m, s_logit[s]);
    red[tid] = m; __syncthreads();
    for (int o = 128; o > 0; o >>= 1) { if (tid < o) red[tid] = fmaxf(red[tid], red[tid + o]); __syncthreads(); }
    m = red[0]; __syncthreads();
    float ssum = 0.f;
    for (int s = tid; s < kS; s += 256) ssum += __expf(s_logit[s] - m);
    red[tid] = ssum; __syncthreads();
    for (int o = 128; o > 0; o >>= 1) { if (tid < o) red[tid] += red[tid + o]; __syncthreads(); }
    const float inv = 1.0f / red[0]; __syncthreads();
    for (int s = tid; s < kS; s += 256) s_attn[s] = __expf(s_logit[s] - m) * inv;
    __syncthreads();
    if (tid < 16) out_attn[(size_t)t * kS + blk * 16 + tid] = s_attn[blk * 16 + tid];

    // S4: ctx slice (16 floats = 4 float4 per block); 64 partials per float4
    {
      const int jq = tid & 3, part = tid >> 2;       // jq: float4 in slice; part: s chunk
      const int col = blk * 4 + jq;                  // float4 column index in enc
      const float4* enc4 = (const float4*)enc;
      float4 acc4 = {0.f, 0.f, 0.f, 0.f};
      const int sb = part * 16;
      for (int s = sb; s < sb + 16; ++s) {
        const float av = s_attn[s];
        const float4 e4 = enc4[(size_t)s * (kH / 4) + col];
        acc4.x += av * e4.x; acc4.y += av * e4.y;
        acc4.z += av * e4.z; acc4.w += av * e4.w;
      }
      red4[tid] = acc4; __syncthreads();
      if (tid < 4) {
        float4 c = red4[tid];
        for (int p = 1; p < 64; ++p) {
          const float4 r = red4[p * 4 + tid];
          c.x += r.x; c.y += r.y; c.z += r.z; c.w += r.w;
        }
        ((float4*)ctx)[blk * 4 + tid] = c;
      }
    }
    grid_barrier(bar, epoch);

    // S5: GRU rows for hidden slice [blk*16, blk*16+16), float4 dots
    {
      const int i0 = blk * 16;
      const float4* ctx4 = (const float4*)ctx;
      for (int lr = wv; lr < 48; lr += 8) {
        const int gate = lr >> 4, ii = lr & 15;
        const int row = gate * kH + i0 + ii;
        const float4* wih4 = (const float4*)(Wih + (size_t)row * (kE + kH));
        const float4* whh4 = (const float4*)(Whh + (size_t)row * kH);
        float gi = 0.f, gh = 0.f;
        for (int k = lane; k < kE / 4; k += 32) gi += dot4(wih4[k], emb4[k]);
        for (int k = lane; k < kH / 4; k += 32) gi += dot4(wih4[kE / 4 + k], ctx4[k]);
        for (int k = lane; k < kH / 4; k += 32) gh += dot4(whh4[k], h4[k]);
        gi = warp_sum(gi); gh = warp_sum(gh);
        if (lane == 0) { s_gi[lr] = gi + bih[row]; s_gh[lr] = gh + bhh[row]; }
      }
      __syncthreads();
      if (tid < 16) {
        const int ig = i0 + tid;
        const float r = 1.f / (1.f + __expf(-(s_gi[tid] + s_gh[tid])));
        const float z = 1.f / (1.f + __expf(-(s_gi[16 + tid] + s_gh[16 + tid])));
        const float n = tanhf(s_gi[32 + tid] + r * s_gh[32 + tid]);
        const float hn = (1.f - z) * n + z * hcur[ig];
        hnext[ig] = hn;
        Hall[(size_t)t * kH + ig] = hn;
      }
    }
    grid_barrier(bar, epoch);
  }
  if (tid < 16) out_hfinal[blk * 16 + tid] = hbuf[blk * 16 + tid];  // kL even -> buf 0
}

// ---------------------------------------------------------------- WMMA output GEMM
// out[t][v] = Hall[t,:] . Wout[v,:] + bout[v];  bf16 WMMA 16x16x32, f32 accum.
__device__ __forceinline__ void cvt4(v16bf& d, int base, float4 f) {
  d[base + 0] = (__bf16)f.x; d[base + 1] = (__bf16)f.y;
  d[base + 2] = (__bf16)f.z; d[base + 3] = (__bf16)f.w;
}

__global__ void k_out_gemm(const float* __restrict__ Hall, const float* __restrict__ Wout,
                           const float* __restrict__ bout, float* __restrict__ out) {
  const int lane = threadIdx.x & 31;
  const int wv = threadIdx.x >> 5;                 // 8 waves: 4(M) x 2(N)
  const int mbase = blockIdx.x * 64 + (wv >> 1) * 16;
  const int nbase = blockIdx.y * 128 + (wv & 1) * 64;
  const int ml = lane & 15;
  const int hi = lane >> 4;                        // lane half selects K sub-range
  const float* arow = Hall + (size_t)(mbase + ml) * kH;

  int ncl[4];
#pragma unroll
  for (int nf = 0; nf < 4; ++nf) {
    const int n = nbase + nf * 16 + ml;
    ncl[nf] = n < kV ? n : kV - 1;                 // clamp loads; mask stores
  }
  v8f acc[4] = {};

  for (int k0 = 0; k0 < kH; k0 += 32) {
    // A fragment: 16x32 bf16. lanes 0-15: K 0..7 & 16..23; lanes 16-31: K 8..15 & 24..31
    v16bf a;
    {
      const float4* p0 = (const float4*)(arow + k0 + hi * 8);
      const float4* p1 = (const float4*)(arow + k0 + hi * 8 + 16);
      cvt4(a, 0, p0[0]);  cvt4(a, 4, p0[1]);
      cvt4(a, 8, p1[0]);  cvt4(a, 12, p1[1]);
    }
    const int kb = k0 + hi * 8;                    // B: lanes 0-15 K 0..7 & 16..23; lanes 16-31 K 8..15 & 24..31
    if (k0 + 32 < kH) __builtin_prefetch(arow + k0 + 32, 0, 1);
#pragma unroll
    for (int nf = 0; nf < 4; ++nf) {
      const float4* q = (const float4*)(Wout + (size_t)ncl[nf] * kH + kb);
      v16bf b;
      cvt4(b, 0, q[0]); cvt4(b, 4, q[1]); cvt4(b, 8, q[4]); cvt4(b, 12, q[5]);
      acc[nf] = __builtin_amdgcn_wmma_f32_16x16x32_bf16(
          false, a, false, b, (short)0, acc[nf], false, false);
    }
  }

#pragma unroll
  for (int nf = 0; nf < 4; ++nf) {
    const int n = nbase + nf * 16 + ml;
    if (n < kV) {
      const float bo = bout[n];
#pragma unroll
      for (int r = 0; r < 8; ++r)
        out[(size_t)(mbase + hi * 8 + r) * kV + n] = acc[nf][r] + bo;
    }
  }
}

// ---------------------------------------------------------------- log-softmax rows
__global__ void k_logsoftmax(float* __restrict__ out) {
  __shared__ float red[256];
  const int tid = threadIdx.x;
  float* row = out + (size_t)blockIdx.x * kV;
  float m = -3.0e38f;
  for (int v = tid; v < kV; v += 256) m = fmaxf(m, row[v]);
  red[tid] = m; __syncthreads();
  for (int o = 128; o > 0; o >>= 1) { if (tid < o) red[tid] = fmaxf(red[tid], red[tid + o]); __syncthreads(); }
  m = red[0]; __syncthreads();
  float s = 0.f;
  for (int v = tid; v < kV; v += 256) s += __expf(row[v] - m);
  red[tid] = s; __syncthreads();
  for (int o = 128; o > 0; o >>= 1) { if (tid < o) red[tid] += red[tid + o]; __syncthreads(); }
  const float lse = m + __logf(red[0]);
  for (int v = tid; v < kV; v += 256) row[v] -= lse;
}

// ---------------------------------------------------------------- launch
extern "C" void kernel_launch(void* const* d_in, const int* in_sizes, int n_in,
                              void* d_out, int out_size, void* d_ws, size_t ws_size,
                              hipStream_t stream) {
  (void)in_sizes; (void)n_in; (void)out_size; (void)ws_size;
  const int*   tokens = (const int*)  d_in[0];
  const float* hidden = (const float*)d_in[1];
  const float* enc    = (const float*)d_in[2];   // (S,1,H) contiguous
  const float* emb    = (const float*)d_in[3];
  const float* W1  = (const float*)d_in[4];  const float* b1  = (const float*)d_in[5];
  const float* W2  = (const float*)d_in[6];  const float* b2  = (const float*)d_in[7];
  const float* W3  = (const float*)d_in[8];  const float* b3  = (const float*)d_in[9];
  const float* Wa  = (const float*)d_in[10]; const float* ba  = (const float*)d_in[11];
  const float* Wih = (const float*)d_in[12]; const float* bih = (const float*)d_in[13];
  const float* Whh = (const float*)d_in[14]; const float* bhh = (const float*)d_in[15];
  const float* Wout = (const float*)d_in[16]; const float* bout = (const float*)d_in[17];

  // workspace layout (all 256B aligned)
  char* w = (char*)d_ws;
  unsigned* bar   = (unsigned*)w;                  // 256 B barrier counter
  float* embedded = (float*)(w + 256);             // L*E
  float* ah2      = embedded + (size_t)kL * kE;    // S*A
  float* ah3      = ah2 + (size_t)kS * kA;         // L*A
  float* hbuf     = ah3 + (size_t)kL * kA;         // 2*H (double buffer)
  float* ctx      = hbuf + 2 * kH;                 // H
  float* Hall     = ctx + kH;                      // L*H

  float* out_logp   = (float*)d_out;               // (L,V)
  float* out_hfinal = out_logp + (size_t)kL * kV;  // (H)
  float* out_attn   = out_hfinal + kH;             // (L,S)

  hipMemsetAsync(d_ws, 0, 256, stream);            // reset grid barrier (capture-safe)

  k_embed<<<kL, 256, 0, stream>>>(tokens, emb, embedded);
  k_proj <<<kS, 256, 0, stream>>>(enc, W2, b2, ah2, kH);
  k_proj <<<kL, 256, 0, stream>>>(embedded, W3, b3, ah3, kE);

  k_scan<<<kG, 256, 0, stream>>>(embedded, enc, hidden, W1, b1, Wa, ba,
                                 Wih, bih, Whh, bhh, ah2, ah3,
                                 hbuf, ctx, Hall,
                                 out_attn, out_hfinal, bar);

  dim3 g((kL + 63) / 64, (kV + 127) / 128);        // 8 x 393
  k_out_gemm<<<g, 256, 0, stream>>>(Hall, Wout, bout, out_logp);

  k_logsoftmax<<<kL, 256, 0, stream>>>(out_logp);
}
